// M9_81071802679815
// MI455X (gfx1250) — compile-verified, matrix operand order checked by reading, not run on hardware
//
#include <hip/hip_runtime.h>
#include <hip/hip_bf16.h>

#define N_NODES 100000
#define N_EDGES 1600000
#define F_IN 64
#define F_HID 128
#define NDEG 7
#define MAXT ((N_NODES + 15) / 16 + NDEG)
#define NB_SCAN ((N_NODES + 255) / 256)

typedef __attribute__((ext_vector_type(16))) __bf16 v16bf;
typedef __attribute__((ext_vector_type(8)))  float v8f;
typedef __attribute__((ext_vector_type(4)))  unsigned int v4u;

union Frag { v4u q[2]; v16bf v; };

__device__ __forceinline__ unsigned short f2bf(float f) {
    unsigned u = __builtin_bit_cast(unsigned, f);
    unsigned r = u + 0x7FFFu + ((u >> 16) & 1u);   // round-to-nearest-even
    return (unsigned short)(r >> 16);
}

// ---------------- utility kernels ----------------
__global__ void k_zero_u32(unsigned* p, size_t n) {
    size_t i = (size_t)blockIdx.x * blockDim.x + threadIdx.x;
    if (i < n) p[i] = 0u;
}

__global__ void k_cvt_bf(const float* __restrict__ a, unsigned short* __restrict__ b, size_t n) {
    size_t i = (size_t)blockIdx.x * blockDim.x + threadIdx.x;
    if (i < n) b[i] = f2bf(a[i]);
}

// W: [7][K][128] fp32 -> Wt: [7][128][K] bf16 (N-major so B fragments load contiguously)
__global__ void k_cvt_wt(const float* __restrict__ W, unsigned short* __restrict__ Wt, int K) {
    size_t i = (size_t)blockIdx.x * blockDim.x + threadIdx.x;
    size_t tot = (size_t)NDEG * K * F_HID;
    if (i >= tot) return;
    int d = (int)(i / ((size_t)K * F_HID));
    int rem = (int)(i % ((size_t)K * F_HID));
    int k = rem / F_HID;
    int n = rem % F_HID;
    Wt[((size_t)d * F_HID + n) * K + k] = f2bf(W[i]);
}

// ---------------- degree ----------------
__global__ void k_deg(const long long* __restrict__ dst, int* __restrict__ deg, int e) {
    int i = blockIdx.x * blockDim.x + threadIdx.x;
    if (i < e) atomicAdd(&deg[(int)dst[i]], 1);
}

// ---------------- CSR build: two-level exclusive scan of (unclamped) degrees ----------------
__global__ void k_block_red(const int* __restrict__ deg, int* __restrict__ bsum, int n) {
    __shared__ int s[256];
    int i = blockIdx.x * 256 + threadIdx.x;
    s[threadIdx.x] = (i < n) ? deg[i] : 0;
    __syncthreads();
    for (int o = 128; o > 0; o >>= 1) {
        if (threadIdx.x < o) s[threadIdx.x] += s[threadIdx.x + o];
        __syncthreads();
    }
    if (threadIdx.x == 0) bsum[blockIdx.x] = s[0];
}

__global__ void k_scan_bsum(int* bsum, int nb) {  // single thread, ~391 iters
    int acc = 0;
    for (int b = 0; b < nb; ++b) { int v = bsum[b]; bsum[b] = acc; acc += v; }
}

__global__ void k_scan_final(const int* __restrict__ deg, const int* __restrict__ bsum,
                             int* __restrict__ rowptr, int* __restrict__ cur, int n) {
    __shared__ int s[256];
    int i = blockIdx.x * 256 + threadIdx.x;
    int v = (i < n) ? deg[i] : 0;
    s[threadIdx.x] = v;
    __syncthreads();
    for (int o = 1; o < 256; o <<= 1) {          // Hillis-Steele inclusive scan
        int t = (threadIdx.x >= o) ? s[threadIdx.x - o] : 0;
        __syncthreads();
        s[threadIdx.x] += t;
        __syncthreads();
    }
    if (i < n) {
        int ex = bsum[blockIdx.x] + s[threadIdx.x] - v;
        rowptr[i] = ex;
        cur[i] = ex;
        if (i == n - 1) rowptr[n] = bsum[blockIdx.x] + s[threadIdx.x];
    }
}

__global__ void k_fill_csr(const long long* __restrict__ src, const long long* __restrict__ dst,
                           int* __restrict__ cur, int* __restrict__ csr, int e) {
    int i = blockIdx.x * blockDim.x + threadIdx.x;
    if (i >= e) return;
    int pos = atomicAdd(&cur[(int)dst[i]], 1);
    csr[pos] = (int)src[i];
}

// ---------------- degree buckets (clamped) ----------------
// small[]: 0..6 cnt, 7..14 bstart(8), 15..21 cursor(7), 22 ntiles
__global__ void k_clamp_count(int* __restrict__ deg, int* __restrict__ cnt, int n) {
    int i = blockIdx.x * blockDim.x + threadIdx.x;
    if (i >= n) return;
    int d = deg[i]; if (d > NDEG - 1) d = NDEG - 1;
    deg[i] = d;
    atomicAdd(&cnt[d], 1);
}

__global__ void k_scan7(int* small) {
    int acc = 0;
    for (int d = 0; d < NDEG; ++d) {
        small[7 + d] = acc;
        small[15 + d] = acc;
        acc += small[d];
    }
    small[7 + NDEG] = acc;
}

__global__ void k_fill_order(const int* __restrict__ deg, int* __restrict__ cur,
                             int* __restrict__ order, int n) {
    int i = blockIdx.x * blockDim.x + threadIdx.x;
    if (i >= n) return;
    int pos = atomicAdd(&cur[deg[i]], 1);
    order[pos] = i;
}

__global__ void k_build_tiles(const int* __restrict__ bs, int* __restrict__ tiles,
                              int* __restrict__ ntiles) {
    int t = 0;
    for (int d = 0; d < NDEG; ++d) {
        for (int base = bs[d]; base < bs[d + 1]; base += 16) {
            int c = bs[d + 1] - base; if (c > 16) c = 16;
            tiles[3 * t] = base; tiles[3 * t + 1] = c; tiles[3 * t + 2] = d; ++t;
        }
    }
    *ntiles = t;
}

// ---------------- CSR gather-reduce (replaces scatter atomics) ----------------
// One wave per node; lane l accumulates features [4l, 4l+4) over the node's edge list
// (contiguous 512B row loads), then packs fp32->bf16 and stores h directly in bf16.
__global__ void k_gather(const float* __restrict__ x, const int* __restrict__ rowptr,
                         const int* __restrict__ csr, unsigned short* __restrict__ hb, int K) {
    int wid = (int)(((size_t)blockIdx.x * blockDim.x + threadIdx.x) >> 5);
    int lane = threadIdx.x & 31;
    if (wid >= N_NODES) return;
    int rs = rowptr[wid], re = rowptr[wid + 1];
    if (lane < (K >> 2)) {
        float4 acc = make_float4(0.f, 0.f, 0.f, 0.f);
        for (int j = rs; j < re; ++j) {
            int s = csr[j];
            if (j + 1 < re)
                __builtin_prefetch(x + (size_t)csr[j + 1] * K + lane * 4, 0, 0);
            float4 v = *(const float4*)(x + (size_t)s * K + lane * 4);
            acc.x += v.x; acc.y += v.y; acc.z += v.z; acc.w += v.w;
        }
        uint2 pv;
        pv.x = ((unsigned)f2bf(acc.y) << 16) | f2bf(acc.x);
        pv.y = ((unsigned)f2bf(acc.w) << 16) | f2bf(acc.z);
        *(uint2*)(hb + (size_t)wid * K + lane * 4) = pv;
    }
}

// ---------------- WMMA per-degree GEMM ----------------
// 256 threads = 8 waves; wave w computes a 16(node)x16(out-col) tile, cols [16w,16w+16).
// A fragment (bf16 16x32): lane holds row M=lane&15; lanes<16 read K {k0..k0+7, k0+16..k0+23},
// lanes>=16 read K {k0+8..k0+15, k0+24..k0+31} -> two 16B loads per fragment.
template <int K>
__launch_bounds__(256)
__global__ void k_mfconv_wmma(const unsigned short* __restrict__ xb,
                              const unsigned short* __restrict__ hb,
                              const unsigned short* __restrict__ Wlt,  // [7][128][K] bf16
                              const unsigned short* __restrict__ Wrt,  // [7][128][K] bf16
                              const float* __restrict__ bl,            // [7][128]
                              const int* __restrict__ order,
                              const int* __restrict__ tiles,
                              const int* __restrict__ ntiles,
                              float* __restrict__ out) {
    int t = blockIdx.x;
    if (t >= *ntiles) return;
    int start = tiles[3 * t], cnt = tiles[3 * t + 1], d = tiles[3 * t + 2];

    __shared__ int nidx[16];
    if (threadIdx.x < 16) {
        int r = threadIdx.x;
        nidx[r] = order[start + (r < cnt ? r : 0)];
    }
    __syncthreads();

    int wave = threadIdx.x >> 5;
    int lane = threadIdx.x & 31;
    int half = lane >> 4;          // which 16-lane half
    int m = lane & 15;             // A row within tile
    int n = (wave << 4) + m;       // output column (also B column)

    const unsigned short* xr = xb + (size_t)nidx[m] * K + half * 8;
    const unsigned short* hr = hb + (size_t)nidx[m] * K + half * 8;
    const unsigned short* wl = Wlt + ((size_t)d * F_HID + n) * K + half * 8;
    const unsigned short* wr = Wrt + ((size_t)d * F_HID + n) * K + half * 8;

    v8f c = {};
#pragma unroll
    for (int k0 = 0; k0 < K; k0 += 32) {
        Frag ax, ah, bw, cw;
        ax.q[0] = *(const v4u*)(xr + k0);  ax.q[1] = *(const v4u*)(xr + k0 + 16);
        ah.q[0] = *(const v4u*)(hr + k0);  ah.q[1] = *(const v4u*)(hr + k0 + 16);
        bw.q[0] = *(const v4u*)(wl + k0);  bw.q[1] = *(const v4u*)(wl + k0 + 16);
        cw.q[0] = *(const v4u*)(wr + k0);  cw.q[1] = *(const v4u*)(wr + k0 + 16);
        c = __builtin_amdgcn_wmma_f32_16x16x32_bf16(false, ax.v, false, bw.v,
                                                    (short)0, c, false, false);
        c = __builtin_amdgcn_wmma_f32_16x16x32_bf16(false, ah.v, false, cw.v,
                                                    (short)0, c, false, false);
    }

    float bias = bl[d * F_HID + n];
#pragma unroll
    for (int v = 0; v < 8; ++v) {
        int row = v + (half << 3);
        if (row < cnt)
            out[(size_t)nidx[row] * F_HID + n] = c[v] + bias;
    }
}

// ---------------- BatchNorm (training-mode) + LeakyReLU ----------------
__global__ void k_bn_stats(const float* __restrict__ y, float* __restrict__ sums, int n) {
    int c = threadIdx.x;  // blockDim.x == 128
    float s = 0.f, q = 0.f;
    for (int r = blockIdx.x; r < n; r += gridDim.x) {
        float v = y[(size_t)r * F_HID + c];
        s += v; q += v * v;
    }
    atomicAdd(&sums[c], s);
    atomicAdd(&sums[F_HID + c], q);
}

__global__ void k_bn_apply(const float* __restrict__ y, const float* __restrict__ sums,
                           const float* __restrict__ g, const float* __restrict__ b,
                           float* __restrict__ xf, unsigned short* __restrict__ xb, int n) {
    size_t i = (size_t)blockIdx.x * blockDim.x + threadIdx.x;
    if (i >= (size_t)n * F_HID) return;
    int c = (int)(i & (F_HID - 1));
    float inv = 1.0f / (float)n;
    float mu = sums[c] * inv;
    float var = sums[F_HID + c] * inv - mu * mu;
    float sc = rsqrtf(var + 1e-5f) * g[c];
    float v = (y[i] - mu) * sc + b[c];
    v = v >= 0.f ? v : 0.01f * v;
    xf[i] = v;
    xb[i] = f2bf(v);
}

// ---------------- final linear 128 -> 2 ----------------
__global__ void k_final(const float* __restrict__ of, const float* __restrict__ Wf,
                        const float* __restrict__ bf_, float* __restrict__ out, int n) {
    int i = blockIdx.x * blockDim.x + threadIdx.x;
    if (i >= n) return;
    float a0 = bf_[0], a1 = bf_[1];
    const float* r = of + (size_t)i * F_HID;
#pragma unroll 8
    for (int k = 0; k < F_HID; ++k) {
        float v = r[k];
        a0 += v * Wf[2 * k];
        a1 += v * Wf[2 * k + 1];
    }
    out[2 * i] = a0;
    out[2 * i + 1] = a1;
}

// ---------------- launch ----------------
extern "C" void kernel_launch(void* const* d_in, const int* in_sizes, int n_in,
                              void* d_out, int out_size, void* d_ws, size_t ws_size,
                              hipStream_t stream) {
    const float*     x   = (const float*)d_in[0];
    const long long* ei  = (const long long*)d_in[1];
    const float* Wl0 = (const float*)d_in[2];
    const float* Wr0 = (const float*)d_in[3];
    const float* bl0 = (const float*)d_in[4];
    const float* Wl1 = (const float*)d_in[5];
    const float* Wr1 = (const float*)d_in[6];
    const float* bl1 = (const float*)d_in[7];
    const float* Wl2 = (const float*)d_in[8];
    const float* Wr2 = (const float*)d_in[9];
    const float* bl2 = (const float*)d_in[10];
    const float* g0  = (const float*)d_in[11];
    const float* b0  = (const float*)d_in[12];
    const float* g1  = (const float*)d_in[13];
    const float* b1  = (const float*)d_in[14];
    const float* Wf  = (const float*)d_in[15];
    const float* bff = (const float*)d_in[16];
    float* outp = (float*)d_out;

    const long long* src = ei;
    const long long* dst = ei + N_EDGES;

    char* ws = (char*)d_ws;
    size_t off = 0;
    auto alloc = [&](size_t bytes) -> char* {
        char* p = ws + off;
        off += (bytes + 255) & ~(size_t)255;
        return p;
    };

    float*          xf    = (float*)alloc((size_t)N_NODES * F_HID * 4);
    float*          of    = (float*)alloc((size_t)N_NODES * F_HID * 4);
    unsigned short* xb    = (unsigned short*)alloc((size_t)N_NODES * F_HID * 2);
    unsigned short* hb    = (unsigned short*)alloc((size_t)N_NODES * F_HID * 2);
    int*            deg   = (int*)alloc((size_t)N_NODES * 4);
    int*            order = (int*)alloc((size_t)N_NODES * 4);
    int*            rowptr= (int*)alloc((size_t)(N_NODES + 1) * 4);
    int*            csrcur= (int*)alloc((size_t)N_NODES * 4);
    int*            csr   = (int*)alloc((size_t)N_EDGES * 4);
    int*            bsum  = (int*)alloc((size_t)NB_SCAN * 4);
    int*            small = (int*)alloc(64 * 4);
    int*            tiles = (int*)alloc((size_t)MAXT * 3 * 4);
    unsigned short* wl0t  = (unsigned short*)alloc((size_t)NDEG * F_IN  * F_HID * 2);
    unsigned short* wr0t  = (unsigned short*)alloc((size_t)NDEG * F_IN  * F_HID * 2);
    unsigned short* wl1t  = (unsigned short*)alloc((size_t)NDEG * F_HID * F_HID * 2);
    unsigned short* wr1t  = (unsigned short*)alloc((size_t)NDEG * F_HID * F_HID * 2);
    unsigned short* wl2t  = (unsigned short*)alloc((size_t)NDEG * F_HID * F_HID * 2);
    unsigned short* wr2t  = (unsigned short*)alloc((size_t)NDEG * F_HID * F_HID * 2);
    float*          bnsum = (float*)alloc(2 * F_HID * 4);

    auto gs = [](size_t n) { return (unsigned)((n + 255) / 256); };

    // ---- degrees ----
    k_zero_u32<<<gs(N_NODES), 256, 0, stream>>>((unsigned*)deg, N_NODES);
    k_zero_u32<<<1, 64, 0, stream>>>((unsigned*)small, 64);
    k_deg<<<gs(N_EDGES), 256, 0, stream>>>(dst, deg, N_EDGES);

    // ---- CSR (uses raw degrees) ----
    k_block_red<<<NB_SCAN, 256, 0, stream>>>(deg, bsum, N_NODES);
    k_scan_bsum<<<1, 1, 0, stream>>>(bsum, NB_SCAN);
    k_scan_final<<<NB_SCAN, 256, 0, stream>>>(deg, bsum, rowptr, csrcur, N_NODES);
    k_fill_csr<<<gs(N_EDGES), 256, 0, stream>>>(src, dst, csrcur, csr, N_EDGES);

    // ---- degree buckets + tiles (clamps deg in place) ----
    k_clamp_count<<<gs(N_NODES), 256, 0, stream>>>(deg, small, N_NODES);
    k_scan7<<<1, 1, 0, stream>>>(small);
    k_fill_order<<<gs(N_NODES), 256, 0, stream>>>(deg, small + 15, order, N_NODES);
    k_build_tiles<<<1, 1, 0, stream>>>(small + 7, tiles, small + 22);

    // ---- weights -> bf16, N-major ----
    k_cvt_wt<<<gs((size_t)NDEG * F_IN  * F_HID), 256, 0, stream>>>(Wl0, wl0t, F_IN);
    k_cvt_wt<<<gs((size_t)NDEG * F_IN  * F_HID), 256, 0, stream>>>(Wr0, wr0t, F_IN);
    k_cvt_wt<<<gs((size_t)NDEG * F_HID * F_HID), 256, 0, stream>>>(Wl1, wl1t, F_HID);
    k_cvt_wt<<<gs((size_t)NDEG * F_HID * F_HID), 256, 0, stream>>>(Wr1, wr1t, F_HID);
    k_cvt_wt<<<gs((size_t)NDEG * F_HID * F_HID), 256, 0, stream>>>(Wl2, wl2t, F_HID);
    k_cvt_wt<<<gs((size_t)NDEG * F_HID * F_HID), 256, 0, stream>>>(Wr2, wr2t, F_HID);

    const unsigned gWave = gs((size_t)N_NODES * 32);  // one wave per node

    // ---- layer 0 (K = 64) ----
    k_cvt_bf<<<gs((size_t)N_NODES * F_IN), 256, 0, stream>>>(x, xb, (size_t)N_NODES * F_IN);
    k_gather<<<gWave, 256, 0, stream>>>(x, rowptr, csr, hb, F_IN);
    k_mfconv_wmma<F_IN><<<MAXT, 256, 0, stream>>>(xb, hb, wl0t, wr0t, bl0, order, tiles,
                                                  small + 22, of);
    k_zero_u32<<<1, 256, 0, stream>>>((unsigned*)bnsum, 2 * F_HID);
    k_bn_stats<<<512, F_HID, 0, stream>>>(of, bnsum, N_NODES);
    k_bn_apply<<<gs((size_t)N_NODES * F_HID), 256, 0, stream>>>(of, bnsum, g0, b0, xf, xb, N_NODES);

    // ---- layer 1 (K = 128) ----
    k_gather<<<gWave, 256, 0, stream>>>(xf, rowptr, csr, hb, F_HID);
    k_mfconv_wmma<F_HID><<<MAXT, 256, 0, stream>>>(xb, hb, wl1t, wr1t, bl1, order, tiles,
                                                   small + 22, of);
    k_zero_u32<<<1, 256, 0, stream>>>((unsigned*)bnsum, 2 * F_HID);
    k_bn_stats<<<512, F_HID, 0, stream>>>(of, bnsum, N_NODES);
    k_bn_apply<<<gs((size_t)N_NODES * F_HID), 256, 0, stream>>>(of, bnsum, g1, b1, xf, xb, N_NODES);

    // ---- layer 2 (K = 128, no BN/act) ----
    k_gather<<<gWave, 256, 0, stream>>>(xf, rowptr, csr, hb, F_HID);
    k_mfconv_wmma<F_HID><<<MAXT, 256, 0, stream>>>(xb, hb, wl2t, wr2t, bl2, order, tiles,
                                                   small + 22, of);

    // ---- final linear ----
    k_final<<<gs(N_NODES), 256, 0, stream>>>(of, Wf, bff, outp, N_NODES);
}
